// StylePredictor_21225728377073
// MI455X (gfx1250) — matrix operand order repaired and checked by reading, not compiled, not run on hardware
//
#include <hip/hip_runtime.h>
#include <math.h>
#include <stdint.h>

typedef __attribute__((ext_vector_type(16))) _Float16 v16h;
typedef __attribute__((ext_vector_type(8)))  float    v8f;

#define DEVINL __device__ __forceinline__

constexpr int Bn = 16, T = 1024, Dm = 256, Hh = 2, DKd = 128, Kc = 5;
constexpr int Mrows = Bn * T;       // 16384
constexpr int CK = Dm * Kc;         // 1280

union AF { v16h v; _Float16 e[16]; };
union CF { v8f  v; float    e[8];  };

DEVINL v8f wmma16(v16h a, v16h b, v8f c) {
  // D = A(16x32 f16) * B(32x16 f16) + C(16x16 f32)
  return __builtin_amdgcn_wmma_f32_16x16x32_f16(false, a, false, b, (short)0, c,
                                                false, false);
}
DEVINL float frcp(float x) { return __builtin_amdgcn_rcpf(x); }
DEVINL float sigm(float x) { return frcp(1.f + __expf(-x)); }
DEVINL float mishf(float x) {
  // x * tanh(softplus(x));  tanh(log u) = (u^2-1)/(u^2+1), u = 1+e^x
  if (x > 15.f) return x;
  float e = __expf(x);
  float u = 1.f + e;
  float u2 = u * u;
  return x * (u2 - 1.f) * frcp(u2 + 1.f);
}

// ---------------------------------------------------------------- f32 -> f16
__global__ void cvt_f16(const float* __restrict__ s, _Float16* __restrict__ d, int n) {
  int i = blockIdx.x * 256 + threadIdx.x;
  if (i < n) d[i] = (_Float16)s[i];
}

// ------------------------- conv weight: f32[oc][ic*5+dk] -> f16[oc][dk*256+ic]
__global__ void cvt_convw(const float* __restrict__ s, _Float16* __restrict__ d) {
  int i = blockIdx.x * 256 + threadIdx.x;          // coalesced output index
  int oc = i / CK, kf = i - oc * CK;
  int dk = kf >> 8, ic = kf & 255;
  d[i] = (_Float16)s[oc * CK + ic * Kc + dk];
}

// ------------------------------------------------------------- mask zeroing
__global__ void maskzero_k(const _Float16* __restrict__ in, const int* __restrict__ mask,
                           _Float16* __restrict__ out) {
  int i = blockIdx.x * 256 + threadIdx.x;          // exact coverage of Mrows*Dm
  int row = i >> 8;
  out[i] = mask[row] ? (_Float16)0.f : in[i];
}

// ------------------------------------------------------ generic WMMA linear
// Y[M=16384, 256] = act(A @ W^T + bias) (+resid) ; one wave = one 16x16 tile.
// ALAY: 0 = A row-major [M,256]; 1 = A in head layout [B,H,T,DK]
// OLAY: 0 = f16 [M,256]; 1 = f16 [B,H,T,DK]; 2 = f16 [B,H,DK,T]; 3 = f32 masked [M,256]
template <int ACT, int ALAY, int OLAY>
__global__ __launch_bounds__(32) void linear_k(
    const _Float16* __restrict__ A, const _Float16* __restrict__ W,
    const float* __restrict__ bias, const _Float16* __restrict__ resid,
    const int* __restrict__ mask, _Float16* __restrict__ out16,
    float* __restrict__ out32, float oscale) {
  const int lane = threadIdx.x;
  const int m0 = blockIdx.x * 16;
  const int n0 = blockIdx.y * 16;
  const int mrow = m0 + (lane & 15);
  const int ncol = n0 + (lane & 15);
  const int koff = (lane >> 4) * 16;
  const int boff = (lane >> 4) * 16;

  CF acc;
#pragma unroll
  for (int i = 0; i < 8; ++i) acc.e[i] = 0.f;

#pragma unroll
  for (int k0 = 0; k0 < Dm; k0 += 32) {
    AF a;
    if (ALAY == 0) {
      const _Float16* p = A + (size_t)mrow * Dm + k0 + koff;
#pragma unroll
      for (int i = 0; i < 8; ++i) { a.e[i] = p[i]; a.e[i + 8] = p[8 + i]; }
    } else {
      const int bb = mrow >> 10, tt = mrow & (T - 1);
      const int k1 = k0 + koff, k2 = k1 + 8;
      const _Float16* p1 = A + (((size_t)(bb * Hh + (k1 >> 7)) * T + tt) << 7) + (k1 & 127);
      const _Float16* p2 = A + (((size_t)(bb * Hh + (k2 >> 7)) * T + tt) << 7) + (k2 & 127);
#pragma unroll
      for (int i = 0; i < 8; ++i) { a.e[i] = p1[i]; a.e[i + 8] = p2[i]; }
    }
    AF bf;
    const _Float16* pw = W + (size_t)ncol * Dm + k0 + boff;
    __builtin_prefetch((const void*)(pw + 64), 0, 0);
#pragma unroll
    for (int i = 0; i < 16; ++i) bf.e[i] = pw[i];
    acc.v = wmma16(a.v, bf.v, acc.v);
  }

  const float bv = bias[ncol];
#pragma unroll
  for (int r = 0; r < 8; ++r) {
    const int row = m0 + r + (lane >> 4) * 8;
    float v = acc.e[r] + bv;
    if (ACT == 1) v = mishf(v);
    if (resid) v += (float)resid[(size_t)row * Dm + ncol];
    v *= oscale;
    if (OLAY == 0) {
      out16[(size_t)row * Dm + ncol] = (_Float16)v;
    } else if (OLAY == 1) {
      const int bb = row >> 10, tt = row & (T - 1);
      out16[(((size_t)(bb * Hh + (ncol >> 7)) * T + tt) << 7) + (ncol & 127)] = (_Float16)v;
    } else if (OLAY == 2) {
      const int bb = row >> 10, tt = row & (T - 1);
      out16[(((size_t)(bb * Hh + (ncol >> 7)) * DKd + (ncol & 127)) << 10) + tt] = (_Float16)v;
    } else {
      out32[(size_t)row * Dm + ncol] = mask[row] ? 0.f : v;
    }
  }
}

// -------------------------------------------------- Conv1dGLU as WMMA GEMM
// Weights pre-permuted to K-order kf = dk*256+ic so A-fragments are contiguous.
// Halo tile staged with gfx1250 async global->LDS DMA (ASYNCcnt).
__global__ __launch_bounds__(256) void conv_glu_k(
    const _Float16* __restrict__ hin, const _Float16* __restrict__ cw,
    const float* __restrict__ cb, _Float16* __restrict__ hout) {
  __shared__ _Float16 sh[20 * Dm];                   // t0-2 .. t0+17 halo tile
  const int t0 = blockIdx.x * 16;
  const int b = blockIdx.y;
  const int tid = threadIdx.x;

  // ---- async staging: 20 rows * 512B = 640 x 16B chunks
#pragma unroll
  for (int pass = 0; pass < 3; ++pass) {
    const int idx = pass * 256 + tid;
    if (idx < 640) {
      const int r = idx >> 5;                        // 32 chunks per row
      const int ch = idx & 31;
      const int t = t0 + r - 2;
      _Float16* lp = &sh[r * Dm + ch * 8];
      if (t >= 0 && t < T) {
        const uint32_t lds_off = (uint32_t)(uintptr_t)lp;
        const uint64_t gaddr =
            (uint64_t)(uintptr_t)(hin + ((size_t)(b * T + t)) * Dm + ch * 8);
        asm volatile("global_load_async_to_lds_b128 %0, %1, off"
                     :: "v"(lds_off), "v"(gaddr) : "memory");
      } else {
#pragma unroll
        for (int i = 0; i < 8; ++i) lp[i] = (_Float16)0.f;  // zero halo rows
      }
    }
  }
  asm volatile("s_wait_asynccnt 0" ::: "memory");
  __syncthreads();

  const int lane = tid & 31;
  const int cbase = (tid >> 5) * 32;                 // wave -> 2 col-tiles
  const int mrow = lane & 15;
  const int koff = (lane >> 4) * 16;
  const int nlane = lane & 15;
  const int boff = (lane >> 4) * 16;

  CF aA0, aG0, aA1, aG1;
#pragma unroll
  for (int i = 0; i < 8; ++i) { aA0.e[i] = 0.f; aG0.e[i] = 0.f; aA1.e[i] = 0.f; aG1.e[i] = 0.f; }

#pragma unroll 1
  for (int k0 = 0; k0 < CK; k0 += 32) {
    // kf = dk*256 + ic ; runs of 8 never cross a dk block
    const int k1 = k0 + koff, k2 = k1 + 8;
    const _Float16* p1 = &sh[(mrow + (k1 >> 8)) * Dm + (k1 & 255)];
    const _Float16* p2 = &sh[(mrow + (k2 >> 8)) * Dm + (k2 & 255)];
    AF a;
#pragma unroll
    for (int i = 0; i < 8; ++i) { a.e[i] = p1[i]; a.e[i + 8] = p2[i]; }

    AF b0, b1, g0, g1;
    const _Float16* w0  = cw + (size_t)(cbase + nlane) * CK + k0 + boff;
    const _Float16* w1  = cw + (size_t)(cbase + 16 + nlane) * CK + k0 + boff;
    const _Float16* wg0 = cw + (size_t)(cbase + 256 + nlane) * CK + k0 + boff;
    const _Float16* wg1 = cw + (size_t)(cbase + 256 + 16 + nlane) * CK + k0 + boff;
#pragma unroll
    for (int i = 0; i < 16; ++i) {
      b0.e[i] = w0[i]; b1.e[i] = w1[i]; g0.e[i] = wg0[i]; g1.e[i] = wg1[i];
    }
    aA0.v = wmma16(a.v, b0.v, aA0.v);
    aA1.v = wmma16(a.v, b1.v, aA1.v);
    aG0.v = wmma16(a.v, g0.v, aG0.v);
    aG1.v = wmma16(a.v, g1.v, aG1.v);
  }

#pragma unroll
  for (int r = 0; r < 8; ++r) {
    const int mr = r + (lane >> 4) * 8;
    const int t = t0 + mr;
    {
      const int c = cbase + nlane;
      float av = aA0.e[r] + cb[c];
      float gv = aG0.e[r] + cb[c + 256];
      float res = (float)sh[(mr + 2) * Dm + c] + av * sigm(gv);
      hout[((size_t)(b * T + t)) * Dm + c] = (_Float16)res;
    }
    {
      const int c = cbase + 16 + nlane;
      float av = aA1.e[r] + cb[c];
      float gv = aG1.e[r] + cb[c + 256];
      float res = (float)sh[(mr + 2) * Dm + c] + av * sigm(gv);
      hout[((size_t)(b * T + t)) * Dm + c] = (_Float16)res;
    }
  }
}

// ------------------------------------------ attention scores + soft(max/exp)
// One wave per (b,h,16 queries). Writes unnormalized exp(S - rowmax) and sums.
__global__ __launch_bounds__(32) void attn_scores_k(
    const _Float16* __restrict__ q, const _Float16* __restrict__ kmat,
    const int* __restrict__ mask, _Float16* __restrict__ P,
    float* __restrict__ sums) {
  __shared__ _Float16 s[16 * T];                     // 32 KB score strip
  const int lane = threadIdx.x;
  const int q0 = blockIdx.x * 16;
  const int h = blockIdx.y, b = blockIdx.z;
  const size_t bh = (size_t)(b * Hh + h);
  const _Float16* qb = q + (bh * T) * DKd;
  const _Float16* kb = kmat + (bh * T) * DKd;

  const int m = lane & 15;
  const int koff = (lane >> 4) * 16;
  AF qa[4];
#pragma unroll
  for (int c = 0; c < 4; ++c) {
    const _Float16* p = qb + (size_t)(q0 + m) * DKd + c * 32 + koff;
#pragma unroll
    for (int i = 0; i < 8; ++i) { qa[c].e[i] = p[i]; qa[c].e[i + 8] = p[8 + i]; }
  }

#pragma unroll 1
  for (int kt = 0; kt < T / 16; ++kt) {
    CF acc;
#pragma unroll
    for (int i = 0; i < 8; ++i) acc.e[i] = 0.f;
    const int key = kt * 16 + (lane & 15);
#pragma unroll
    for (int c = 0; c < 4; ++c) {
      AF bf;
      const _Float16* p = kb + (size_t)key * DKd + c * 32 + (lane >> 4) * 16;
#pragma unroll
      for (int i = 0; i < 16; ++i) bf.e[i] = p[i];
      acc.v = wmma16(qa[c].v, bf.v, acc.v);
    }
    const bool mk = mask[b * T + key] != 0;
#pragma unroll
    for (int r = 0; r < 8; ++r) {
      float v = mk ? -__builtin_inff() : acc.e[r];
      s[(r + (lane >> 4) * 8) * T + key] = (_Float16)v;
    }
  }
  __syncthreads();

  const int row = lane & 15, seg = lane >> 4;
  const int base = row * T + seg * 512;
  float mx = -__builtin_inff();
#pragma unroll 1
  for (int j = 0; j < 512; ++j) mx = fmaxf(mx, (float)s[base + j]);
  mx = fmaxf(mx, __shfl_xor(mx, 16, 32));

  float sum = 0.f;
  _Float16* Prow = P + (bh * T + q0 + row) * (size_t)T + seg * 512;
#pragma unroll 1
  for (int j = 0; j < 512; ++j) {
    float e = __expf((float)s[base + j] - mx);
    Prow[j] = (_Float16)e;
    sum += e;
  }
  sum += __shfl_xor(sum, 16, 32);
  if (lane < 16) sums[bh * T + q0 + row] = sum;
}

// ----------------------------------------------------- attention P @ V (WMMA)
// One wave per (b,h,16 queries), full d=128 (8 N-tiles). V pre-transposed [B,H,DK,T].
__global__ __launch_bounds__(32) void attn_pv_k(
    const _Float16* __restrict__ P, const _Float16* __restrict__ vt,
    const float* __restrict__ sums, _Float16* __restrict__ ao) {
  const int lane = threadIdx.x;
  const int q0 = blockIdx.x * 16;
  const int h = blockIdx.y, b = blockIdx.z;
  const size_t bh = (size_t)(b * Hh + h);

  CF acc[8];
#pragma unroll
  for (int t = 0; t < 8; ++t)
#pragma unroll
    for (int i = 0; i < 8; ++i) acc[t].e[i] = 0.f;

  const int m = lane & 15;
  const int koff = (lane >> 4) * 16;
  const _Float16* Pb = P + (bh * T + q0 + m) * (size_t)T;

#pragma unroll 1
  for (int k0 = 0; k0 < T; k0 += 32) {
    AF a;
    const _Float16* p = Pb + k0 + koff;
#pragma unroll
    for (int i = 0; i < 8; ++i) { a.e[i] = p[i]; a.e[i + 8] = p[8 + i]; }
#pragma unroll
    for (int t = 0; t < 8; ++t) {
      const int d = t * 16 + (lane & 15);
      const _Float16* pv = vt + (bh * DKd + d) * (size_t)T + k0 + (lane >> 4) * 16;
      AF bf;
#pragma unroll
      for (int i = 0; i < 16; ++i) bf.e[i] = pv[i];
      acc[t].v = wmma16(a.v, bf.v, acc[t].v);
    }
  }

  float inv[8];
#pragma unroll
  for (int r = 0; r < 8; ++r)
    inv[r] = frcp(sums[bh * T + q0 + r + (lane >> 4) * 8]);
#pragma unroll
  for (int t = 0; t < 8; ++t)
#pragma unroll
    for (int r = 0; r < 8; ++r) {
      const int qrow = q0 + r + (lane >> 4) * 8;
      ao[(bh * T + qrow) * (size_t)DKd + t * 16 + (lane & 15)] =
          (_Float16)(acc[t].e[r] * inv[r]);
    }
}

// --------------------------------------------- masked temporal average pool
__global__ __launch_bounds__(256) void pool_k(const float* __restrict__ yf,
                                              const int* __restrict__ mask,
                                              float* __restrict__ out) {
  __shared__ float slen;
  const int b = blockIdx.x, d = threadIdx.x;
  if (d == 0) {
    int c = 0;
    for (int t = 0; t < T; ++t) c += (mask[b * T + t] == 0);
    slen = (float)c;
  }
  __syncthreads();
  float s = 0.f;
#pragma unroll 1
  for (int t = 0; t < T; ++t) s += yf[((size_t)(b * T + t)) * Dm + d];
  out[b * Dm + d] = s / slen;
}

// ===========================================================================
extern "C" void kernel_launch(void* const* d_in, const int* in_sizes, int n_in,
                              void* d_out, int out_size, void* d_ws, size_t ws_size,
                              hipStream_t stream) {
  (void)in_sizes; (void)n_in; (void)out_size; (void)ws_size;
  const float* x    = (const float*)d_in[0];
  const int*   mask = (const int*)d_in[1];
  const float* W1   = (const float*)d_in[2];
  const float* b1   = (const float*)d_in[3];
  const float* W2   = (const float*)d_in[4];
  const float* b2   = (const float*)d_in[5];
  const float* cW1  = (const float*)d_in[6];
  const float* cb1  = (const float*)d_in[7];
  const float* cW2  = (const float*)d_in[8];
  const float* cb2  = (const float*)d_in[9];
  const float* Wq   = (const float*)d_in[10];
  const float* bq   = (const float*)d_in[11];
  const float* Wk   = (const float*)d_in[12];
  const float* bk   = (const float*)d_in[13];
  const float* Wv   = (const float*)d_in[14];
  const float* bvv  = (const float*)d_in[15];
  const float* Wo   = (const float*)d_in[16];
  const float* bo   = (const float*)d_in[17];
  const float* Wf   = (const float*)d_in[18];
  const float* bfw  = (const float*)d_in[19];

  char* wsp = (char*)d_ws;
  auto carve = [&](size_t bytes) -> char* {
    char* p = wsp;
    wsp += (bytes + 255) & ~(size_t)255;
    return p;
  };

  const size_t NW = (size_t)Dm * Dm;             // 65536
  const size_t NC = (size_t)2 * Dm * Dm * Kc;    // 655360
  const size_t NS = (size_t)Mrows * Dm;          // 4194304

  _Float16* w1h  = (_Float16*)carve(NW * 2);
  _Float16* w2h  = (_Float16*)carve(NW * 2);
  _Float16* wqh  = (_Float16*)carve(NW * 2);
  _Float16* wkh  = (_Float16*)carve(NW * 2);
  _Float16* wvh  = (_Float16*)carve(NW * 2);
  _Float16* woh  = (_Float16*)carve(NW * 2);
  _Float16* wfh  = (_Float16*)carve(NW * 2);
  _Float16* cw1h = (_Float16*)carve(NC * 2);
  _Float16* cw2h = (_Float16*)carve(NC * 2);
  float*    sums = (float*)carve((size_t)Bn * Hh * T * 4);
  _Float16* S0   = (_Float16*)carve(NS * 2);     // x16 -> ao
  _Float16* S1   = (_Float16*)carve(NS * 2);     // h1 -> c2 -> k
  _Float16* S2   = (_Float16*)carve(NS * 2);     // h2 -> hm (residual)
  _Float16* S3   = (_Float16*)carve(NS * 2);     // c1 -> q -> h3
  _Float16* S4   = (_Float16*)carve(NS * 2);     // vT
  _Float16* P    = (_Float16*)carve((size_t)Bn * Hh * T * T * 2);  // 64 MB
  float*    yf   = (float*)P;                    // reuse P region (16 MB need)

  // --- weight/input conversion to f16 (conv weights permuted to dk-major K)
  cvt_f16<<<(int)((NW + 255) / 256), 256, 0, stream>>>(W1, w1h, (int)NW);
  cvt_f16<<<(int)((NW + 255) / 256), 256, 0, stream>>>(W2, w2h, (int)NW);
  cvt_f16<<<(int)((NW + 255) / 256), 256, 0, stream>>>(Wq, wqh, (int)NW);
  cvt_f16<<<(int)((NW + 255) / 256), 256, 0, stream>>>(Wk, wkh, (int)NW);
  cvt_f16<<<(int)((NW + 255) / 256), 256, 0, stream>>>(Wv, wvh, (int)NW);
  cvt_f16<<<(int)((NW + 255) / 256), 256, 0, stream>>>(Wo, woh, (int)NW);
  cvt_f16<<<(int)((NW + 255) / 256), 256, 0, stream>>>(Wf, wfh, (int)NW);
  cvt_convw<<<(int)((NC + 255) / 256), 256, 0, stream>>>(cW1, cw1h);
  cvt_convw<<<(int)((NC + 255) / 256), 256, 0, stream>>>(cW2, cw2h);
  cvt_f16<<<(int)((NS + 255) / 256), 256, 0, stream>>>(x, S0, (int)NS);

  const dim3 lg(Mrows / 16, Dm / 16);            // (1024, 16) tiles
  const dim3 cg(T / 16, Bn);                     // conv blocks
  const dim3 ag(T / 16, Hh, Bn);                 // attention blocks

  // spectral MLP
  linear_k<1, 0, 0><<<lg, 32, 0, stream>>>(S0, w1h, b1, nullptr, nullptr, S1, nullptr, 1.f);
  linear_k<1, 0, 0><<<lg, 32, 0, stream>>>(S1, w2h, b2, nullptr, nullptr, S2, nullptr, 1.f);
  // temporal Conv1dGLU x2
  conv_glu_k<<<cg, 256, 0, stream>>>(S2, cw1h, cb1, S3);
  conv_glu_k<<<cg, 256, 0, stream>>>(S3, cw2h, cb2, S1);
  // zero padded frames (also the attention residual)
  maskzero_k<<<(int)(NS / 256), 256, 0, stream>>>(S1, mask, S2);
  // q/k/v projections (q pre-scaled by 1/sqrt(D)=1/16; v written transposed)
  linear_k<0, 0, 1><<<lg, 32, 0, stream>>>(S2, wqh, bq, nullptr, nullptr, S3, nullptr, 0.0625f);
  linear_k<0, 0, 1><<<lg, 32, 0, stream>>>(S2, wkh, bk, nullptr, nullptr, S1, nullptr, 1.f);
  linear_k<0, 0, 2><<<lg, 32, 0, stream>>>(S2, wvh, bvv, nullptr, nullptr, S4, nullptr, 1.f);
  // attention
  attn_scores_k<<<ag, 32, 0, stream>>>(S3, S1, mask, P, sums);
  attn_pv_k<<<ag, 32, 0, stream>>>(P, S4, sums, S0);
  // output projection + residual
  linear_k<0, 1, 0><<<lg, 32, 0, stream>>>(S0, woh, bo, S2, nullptr, S3, nullptr, 1.f);
  // final fc with masked f32 output, then masked mean pool
  linear_k<0, 0, 3><<<lg, 32, 0, stream>>>(S3, wfh, bfw, nullptr, mask, nullptr, yf, 1.f);
  pool_k<<<Bn, 256, 0, stream>>>(yf, mask, (float*)d_out);
}
